// Qwen3Attention_16853451670158
// MI455X (gfx1250) — matrix operand order repaired and checked by reading, not compiled, not run on hardware
//
#include <hip/hip_runtime.h>
#include <hip/hip_bf16.h>

// ---------------------------------------------------------------------------
// Types for WMMA fragments (gfx1250, wave32)
// ---------------------------------------------------------------------------
typedef _Float16 v16h __attribute__((ext_vector_type(16)));
typedef _Float16 v8h  __attribute__((ext_vector_type(8)));
typedef float    v8f  __attribute__((ext_vector_type(8)));
typedef float    v4f  __attribute__((ext_vector_type(4)));
typedef int      v4i  __attribute__((ext_vector_type(4)));

static __device__ __forceinline__ v8f wmma_f16(v16h a, v16h b, v8f c) {
    // v_wmma_f32_16x16x32_f16  D = A(16x32 f16) * B(32x16 f16) + C(16x16 f32)
    return __builtin_amdgcn_wmma_f32_16x16x32_f16(
        /*neg_a=*/false, a, /*neg_b=*/false, b,
        /*c_mod=*/(short)0, c, /*reuse_a=*/false, /*reuse_b=*/false);
}

// Scheduling fence: force fragment loads to materialize (distinct registers)
// before the WMMA chain, so the chain runs with a single s_wait_dscnt.
#if __has_builtin(__builtin_amdgcn_sched_barrier)
#define SCHED_FENCE() __builtin_amdgcn_sched_barrier(0)
#else
#define SCHED_FENCE() asm volatile("" ::: "memory")
#endif

#define H_HEADS   32
#define KV_HEADS  8
#define DHEAD     128
#define HID       4096
#define QKV_N     6144            // (H + 2*KV) * D
#define V_OFF     5120            // (H + KV) * D
#define ATTN_SCALE 0.08838834764831845f   // 1/sqrt(128)

// ---------------------------------------------------------------------------
// Kernel 0: f32 -> f16 conversion (hidden states)
// ---------------------------------------------------------------------------
__global__ __launch_bounds__(256) void cvt_f32_f16(const float* __restrict__ x,
                                                   _Float16* __restrict__ y, int n) {
    int i = (blockIdx.x * 256 + threadIdx.x) * 4;
    if (i + 3 < n) {
        v4f a = *(const v4f*)(x + i);
        y[i + 0] = (_Float16)a[0];
        y[i + 1] = (_Float16)a[1];
        y[i + 2] = (_Float16)a[2];
        y[i + 3] = (_Float16)a[3];
    }
}

// ---------------------------------------------------------------------------
// Kernel 1/4: C[M][N] = A_f16[M][K] * B_f32[N][K]^T  (both K-contiguous)
// 128x128 block tile, 8 waves; each wave: 16 rows x 128 cols.
// K staged 64-wide in LDS (32 KB); 16 WMMA per staging round; B fragments are
// preloaded into distinct registers (sched fence) ahead of each WMMA chain.
// ---------------------------------------------------------------------------
__global__ __launch_bounds__(256) void wmma_gemm(const _Float16* __restrict__ A,
                                                 const float* __restrict__ B,
                                                 float* __restrict__ C,
                                                 int M, int N, int K) {
    __shared__ __align__(32) _Float16 As[128 * 64];
    __shared__ __align__(32) _Float16 Bs[128 * 64];

    const int tid  = threadIdx.x;
    const int lane = tid & 31;
    const int wave = tid >> 5;
    const int ml   = lane & 15;          // fragment row/col within 16
    const int grp  = lane >> 4;          // half-wave group
    const int koff = grp * 8;            // unused; A frag uses kb as well
    const int kb   = grp * 16;           // B-frag K sub-offset (halves)
    const int m0   = blockIdx.y * 128;
    const int n0   = blockIdx.x * 128;

    const int row  = tid >> 1;           // 0..127
    const int seg  = (tid & 1) * 32;     // halves / floats within 64-wide tile
    const size_t abase = (size_t)(m0 + row) * K + seg;
    const size_t bbase = (size_t)(n0 + row) * K + seg;

    v8f acc[8] = {};

    for (int k0 = 0; k0 < K; k0 += 64) {
        // ---- stage A (f16, 128x64) into LDS: 16 halves per thread
        {
            v4i a0 = *(const v4i*)(A + abase + k0);
            v4i a1 = *(const v4i*)(A + abase + k0 + 16);
            *(v4i*)(As + row * 64 + seg)      = a0;
            *(v4i*)(As + row * 64 + seg + 16) = a1;
        }
        // ---- stage B (f32 -> f16, 128x64) into LDS: 32 floats per thread
        {
            const float* bp = B + bbase + k0;
#pragma unroll
            for (int c = 0; c < 2; ++c) {
                v4f b0 = *(const v4f*)(bp + c * 16);
                v4f b1 = *(const v4f*)(bp + c * 16 + 4);
                v4f b2 = *(const v4f*)(bp + c * 16 + 8);
                v4f b3 = *(const v4f*)(bp + c * 16 + 12);
                v8h h0, h1;
#pragma unroll
                for (int i = 0; i < 4; ++i) {
                    h0[i]     = (_Float16)b0[i];
                    h0[4 + i] = (_Float16)b1[i];
                    h1[i]     = (_Float16)b2[i];
                    h1[4 + i] = (_Float16)b3[i];
                }
                *(v8h*)(Bs + row * 64 + seg + c * 16)     = h0;
                *(v8h*)(Bs + row * 64 + seg + c * 16 + 8) = h1;
            }
        }
        if (k0 + 64 < K) {   // prefetch next K tile (global_prefetch_b8)
            __builtin_prefetch(A + abase + k0 + 64, 0, 3);
            __builtin_prefetch(B + bbase + k0 + 64, 0, 3);
        }
        __syncthreads();

#pragma unroll
        for (int kk = 0; kk < 2; ++kk) {
            // A fragment for this wave's 16-row strip
            const _Float16* ap = As + (wave * 16 + ml) * 64 + kk * 32;
            v8h alo = *(const v8h*)(ap + kb);
            v8h ahi = *(const v8h*)(ap + kb + 8);
            v16h afrag;
#pragma unroll
            for (int i = 0; i < 8; ++i) { afrag[i] = alo[i]; afrag[8 + i] = ahi[i]; }

            // preload all 8 B fragments into live registers...
            v16h bf[8];
#pragma unroll
            for (int nb = 0; nb < 8; ++nb)
                bf[nb] = *(const v16h*)(Bs + (nb * 16 + ml) * 64 + kk * 32 + kb);
            SCHED_FENCE();   // ...then run the WMMA chain back-to-back
#pragma unroll
            for (int nb = 0; nb < 8; ++nb)
                acc[nb] = wmma_f16(afrag, bf[nb], acc[nb]);
        }
        __syncthreads();
    }

    // store: C row = m0 + wave*16 + grp*8 + j, col = n0 + nb*16 + ml
#pragma unroll
    for (int nb = 0; nb < 8; ++nb) {
#pragma unroll
        for (int j = 0; j < 8; ++j) {
            C[(size_t)(m0 + wave * 16 + grp * 8 + j) * N + n0 + nb * 16 + ml] = acc[nb][j];
        }
    }
}

// ---------------------------------------------------------------------------
// Kernel 2: RMSNorm + RoPE on q and k slices of qkv; emit f16 [head][s][d]
// One block per sequence position; wave w handles q heads {w, w+8, w+16, w+24}
// and k head w. Lane l owns d = {l, l+32, l+64, l+96} so RoPE pairs are local.
// ---------------------------------------------------------------------------
__global__ __launch_bounds__(256) void norm_rope(const float* __restrict__ qkv,
                                                 const long long* __restrict__ pos,
                                                 const float* __restrict__ qw,
                                                 const float* __restrict__ kw,
                                                 _Float16* __restrict__ qo,   // [H][S][D]
                                                 _Float16* __restrict__ ko,   // [KV][S][D]
                                                 int S) {
    const int s    = blockIdx.x;
    const int lane = threadIdx.x & 31;
    const int wave = threadIdx.x >> 5;

    const float p  = (float)pos[s];
    // inv_freq[i] = THETA^(-i/64); lane covers i = lane and i = lane+32
    const float f0 = p * powf(1.0e6f, -(float)lane * (1.0f / 64.0f));
    const float f1 = p * powf(1.0e6f, -(float)(lane + 32) * (1.0f / 64.0f));
    const float c0 = cosf(f0), s0 = sinf(f0);
    const float c1 = cosf(f1), s1 = sinf(f1);

#pragma unroll
    for (int t = 0; t < 5; ++t) {
        const bool isq  = (t < 4);
        const int  head = isq ? (wave + t * 8) : wave;
        const float* src = qkv + (size_t)s * QKV_N + (isq ? head * DHEAD : HID + head * DHEAD);

        float x[4];
#pragma unroll
        for (int j = 0; j < 4; ++j) x[j] = src[lane + j * 32];

        float ss = x[0] * x[0] + x[1] * x[1] + x[2] * x[2] + x[3] * x[3];
#pragma unroll
        for (int m = 16; m >= 1; m >>= 1) ss += __shfl_xor(ss, m, 32);
        const float r = rsqrtf(ss * (1.0f / 128.0f) + 1e-6f);

        const float* w = isq ? qw : kw;
#pragma unroll
        for (int j = 0; j < 4; ++j) x[j] = x[j] * r * w[lane + j * 32];

        const float o0 = x[0] * c0 - x[2] * s0;   // d = lane
        const float o2 = x[2] * c0 + x[0] * s0;   // d = lane + 64
        const float o1 = x[1] * c1 - x[3] * s1;   // d = lane + 32
        const float o3 = x[3] * c1 + x[1] * s1;   // d = lane + 96

        _Float16* dst = (isq ? qo : ko) + ((size_t)head * S + s) * DHEAD;
        dst[lane]      = (_Float16)o0;
        dst[lane + 32] = (_Float16)o1;
        dst[lane + 64] = (_Float16)o2;
        dst[lane + 96] = (_Float16)o3;
    }
}

// ---------------------------------------------------------------------------
// Kernel 3: causal flash attention, one block per (128-row q block, head).
// 8 waves; wave w owns q rows [qb*128 + w*16, +16). Key tiles of 32, online
// softmax, WMMA for both Q@K^T and P@V (V transposed into LDS from f32 qkv).
// B fragments preloaded into live registers (sched fence) before WMMA chains.
// ---------------------------------------------------------------------------
__global__ __launch_bounds__(256) void flash_attn(const _Float16* __restrict__ q, // [H][S][D]
                                                  const _Float16* __restrict__ k, // [KV][S][D]
                                                  const float* __restrict__ qkv,  // for V
                                                  _Float16* __restrict__ attn,    // [S][H*D]
                                                  int S) {
    __shared__ __align__(32) _Float16 Ks[32 * 128];   // [t][d]
    __shared__ __align__(32) _Float16 Vt[128 * 32];   // [d][t]
    __shared__ __align__(32) _Float16 Ps[8 * 16 * 32];// per-wave P staging

    const int h    = blockIdx.y;
    const int qb   = blockIdx.x;
    const int kvh  = h >> 2;                  // H/KV = 4
    const int tid  = threadIdx.x;
    const int lane = tid & 31;
    const int wave = tid >> 5;
    const int ml   = lane & 15;
    const int grp  = lane >> 4;
    const int koff = grp * 8;
    const int kb   = grp * 16;
    const int sbase = qb * 128 + wave * 16;

    // Q fragments (A operand over full D=128): 4 x (16x32)
    v16h qf[4];
    {
        const _Float16* qp = q + ((size_t)h * S + (sbase + ml)) * DHEAD;
#pragma unroll
        for (int kk = 0; kk < 4; ++kk) {
            v8h lo = *(const v8h*)(qp + kk * 32 + kb);
            v8h hi = *(const v8h*)(qp + kk * 32 + kb + 8);
#pragma unroll
            for (int i = 0; i < 8; ++i) { qf[kk][i] = lo[i]; qf[kk][8 + i] = hi[i]; }
        }
    }

    v8f oacc[8] = {};
    float rmax[8], rsum[8];
#pragma unroll
    for (int j = 0; j < 8; ++j) { rmax[j] = -3.0e38f; rsum[j] = 0.0f; }

    const int tend = qb * 128 + 128;
    for (int t0 = 0; t0 < tend; t0 += 32) {
        // stage K tile (32 x 128 f16)
        {
            const int trow = tid >> 3, tseg = (tid & 7) * 16;
            v4i kv = *(const v4i*)(k + ((size_t)kvh * S + t0 + trow) * DHEAD + tseg);
            *(v4i*)(Ks + trow * 128 + tseg) = kv;
        }
        // stage V tile transposed (f32 -> f16): Vt[d][t]
        {
            const int tl = tid >> 3, dseg = (tid & 7) * 16;
            const float* vp = qkv + (size_t)(t0 + tl) * QKV_N + V_OFF + kvh * DHEAD + dseg;
            v4f a = *(const v4f*)(vp);
            v4f b = *(const v4f*)(vp + 4);
            v4f c = *(const v4f*)(vp + 8);
            v4f d = *(const v4f*)(vp + 12);
#pragma unroll
            for (int i = 0; i < 4; ++i) {
                Vt[(dseg + i) * 32 + tl]      = (_Float16)a[i];
                Vt[(dseg + 4 + i) * 32 + tl]  = (_Float16)b[i];
                Vt[(dseg + 8 + i) * 32 + tl]  = (_Float16)c[i];
                Vt[(dseg + 12 + i) * 32 + tl] = (_Float16)d[i];
            }
        }
        __syncthreads();

        // scores: S = Q (16x128) @ K^T -> 16x32 (two 16x16 WMMA columns)
        // preload all 8 K fragments first, then run both WMMA chains
        v16h kbf[8];
#pragma unroll
        for (int nb = 0; nb < 2; ++nb)
#pragma unroll
            for (int kk = 0; kk < 4; ++kk)
                kbf[nb * 4 + kk] =
                    *(const v16h*)(Ks + (nb * 16 + ml) * 128 + kk * 32 + kb);
        SCHED_FENCE();

        v8f sc[2];
#pragma unroll
        for (int nb = 0; nb < 2; ++nb) {
            v8f a = {};
#pragma unroll
            for (int kk = 0; kk < 4; ++kk)
                a = wmma_f16(qf[kk], kbf[nb * 4 + kk], a);
            sc[nb] = a;
        }

        // scale + causal mask + online softmax (f32, native v_exp_f32)
#pragma unroll
        for (int j = 0; j < 8; ++j) {
            const int srow = sbase + grp * 8 + j;
            float v0 = sc[0][j] * ATTN_SCALE;
            float v1 = sc[1][j] * ATTN_SCALE;
            if (t0 + ml > srow)      v0 = -1.0e30f;
            if (t0 + 16 + ml > srow) v1 = -1.0e30f;

            float m = fmaxf(v0, v1);
#pragma unroll
            for (int d = 8; d >= 1; d >>= 1) m = fmaxf(m, __shfl_xor(m, d, 32));
            const float mn   = fmaxf(rmax[j], m);
            const float corr = __expf(rmax[j] - mn);
            const float p0   = __expf(v0 - mn);
            const float p1   = __expf(v1 - mn);
            float psum = p0 + p1;
#pragma unroll
            for (int d = 8; d >= 1; d >>= 1) psum += __shfl_xor(psum, d, 32);
            rsum[j] = rsum[j] * corr + psum;
            rmax[j] = mn;
#pragma unroll
            for (int db = 0; db < 8; ++db) oacc[db][j] *= corr;

            // P (C-layout) -> LDS so we can re-read it in A-layout
            _Float16* pp = Ps + (wave * 16 + grp * 8 + j) * 32;
            pp[ml]      = (_Float16)p0;
            pp[16 + ml] = (_Float16)p1;
        }
        // same-wave LDS RAW: ensure all ds stores land before frag reads
        asm volatile("s_wait_dscnt 0" ::: "memory");

        // P fragment (A operand, 16x32)
        v16h pf;
        {
            const _Float16* pp = Ps + (wave * 16 + ml) * 32;
            v8h lo = *(const v8h*)(pp + kb);
            v8h hi = *(const v8h*)(pp + kb + 8);
#pragma unroll
            for (int i = 0; i < 8; ++i) { pf[i] = lo[i]; pf[8 + i] = hi[i]; }
        }

        // O += P (16x32) @ V (32x128): preload V fragments, then WMMA chain
        v16h vbf[8];
#pragma unroll
        for (int db = 0; db < 8; ++db)
            vbf[db] = *(const v16h*)(Vt + (db * 16 + ml) * 32 + kb);
        SCHED_FENCE();
#pragma unroll
        for (int db = 0; db < 8; ++db)
            oacc[db] = wmma_f16(pf, vbf[db], oacc[db]);

        __syncthreads();
    }

    // finalize: O /= l ; write f16 attn[s][h*128 + d]
#pragma unroll
    for (int j = 0; j < 8; ++j) {
        const float inv = 1.0f / rsum[j];
        const int srow = sbase + grp * 8 + j;
        _Float16* op = attn + (size_t)srow * (H_HEADS * DHEAD) + h * DHEAD;
#pragma unroll
        for (int db = 0; db < 8; ++db)
            op[db * 16 + ml] = (_Float16)(oacc[db][j] * inv);
    }
}

// ---------------------------------------------------------------------------
// Launcher
// ---------------------------------------------------------------------------
extern "C" void kernel_launch(void* const* d_in, const int* in_sizes, int n_in,
                              void* d_out, int out_size, void* d_ws, size_t ws_size,
                              hipStream_t stream) {
    const float*     hidden    = (const float*)d_in[0];
    const long long* positions = (const long long*)d_in[1];   // jnp.int64
    const float*     Wqkv      = (const float*)d_in[2];
    const float*     Wo        = (const float*)d_in[3];
    const float*     qnw       = (const float*)d_in[4];
    const float*     knw       = (const float*)d_in[5];
    float*           out       = (float*)d_out;

    const int S = in_sizes[0] / HID;   // 2048

    // workspace layout (bytes):
    //   hid16  : S*HID*2           (16 MB)
    //   qkv    : S*QKV_N*4         (48 MB)
    //   q16    : H*S*D*2           (16 MB)
    //   k16    : KV*S*D*2          ( 4 MB)
    //   attn16 : S*HID*2           (16 MB)
    char* ws = (char*)d_ws;
    _Float16* hid16 = (_Float16*)ws;
    float*    qkv   = (float*)(ws + (size_t)S * HID * 2);
    _Float16* q16   = (_Float16*)(ws + (size_t)S * HID * 2 + (size_t)S * QKV_N * 4);
    _Float16* k16   = q16 + (size_t)H_HEADS * S * DHEAD;
    _Float16* attn  = k16 + (size_t)KV_HEADS * S * DHEAD;

    // 0) hidden f32 -> f16
    cvt_f32_f16<<<(S * HID / 4 + 255) / 256, 256, 0, stream>>>(hidden, hid16, S * HID);

    // 1) qkv = hidden @ Wqkv^T   (M=S, N=6144, K=4096)
    wmma_gemm<<<dim3(QKV_N / 128, S / 128), 256, 0, stream>>>(hid16, Wqkv, qkv, S, QKV_N, HID);

    // 2) RMSNorm + RoPE -> q16 [H][S][D], k16 [KV][S][D]
    norm_rope<<<S, 256, 0, stream>>>(qkv, positions, qnw, knw, q16, k16, S);

    // 3) causal flash attention -> attn16 [S][H*D]
    flash_attn<<<dim3(S / 128, H_HEADS), 256, 0, stream>>>(q16, k16, qkv, attn, S);

    // 4) out = attn @ Wo^T   (M=S, N=4096, K=4096)
    wmma_gemm<<<dim3(HID / 128, S / 128), 256, 0, stream>>>(attn, Wo, out, S, HID, HID);
}
